// GCN_90555090469651
// MI455X (gfx1250) — matrix operand order repaired and checked by reading, not compiled, not run on hardware
//
#include <hip/hip_runtime.h>
#include <hip/hip_bf16.h>
#include <stdint.h>

#define Nn   50000
#define Ee   800000
#define DHID 128
#define DOUT 64
#define EPSS 1e-5f

typedef __attribute__((ext_vector_type(16))) __bf16 v16bf;
typedef __attribute__((ext_vector_type(2)))  __bf16 v2bf;
typedef __attribute__((ext_vector_type(8)))  float  v8f;

__device__ inline __bf16 f2bf(float f) {
  union { float f; uint32_t u; } v; v.f = f;
  uint32_t u = v.u + 0x7FFFu + ((v.u >> 16) & 1u);  // round-to-nearest-even
  union { unsigned short s; __bf16 b; } r;
  r.s = (unsigned short)(u >> 16);
  return r.b;
}

// Convert 8 f32 -> 8 bf16 into a[off..off+7]; use packed HW convert if present.
__device__ inline void cvt8_bf16(const v8f& f, v16bf& a, int off) {
#if __has_builtin(__builtin_amdgcn_cvt_pk_bf16_f32)
#pragma unroll
  for (int i = 0; i < 4; ++i) {
    v2bf p = __builtin_amdgcn_cvt_pk_bf16_f32(f[2 * i], f[2 * i + 1]);
    a[off + 2 * i]     = p[0];
    a[off + 2 * i + 1] = p[1];
  }
#else
#pragma unroll
  for (int i = 0; i < 8; ++i) a[off + i] = f2bf(f[i]);
#endif
}

// ---------------- degree / norm precompute ----------------
__global__ void k_deg_init(float* __restrict__ deg) {
  int i = blockIdx.x * blockDim.x + threadIdx.x;
  if (i < Nn) deg[i] = 1.0f;  // self-loop weight 1
}

__global__ void k_deg_acc(float* __restrict__ deg, const int* __restrict__ dst,
                          const float* __restrict__ ew) {
  int e = blockIdx.x * blockDim.x + threadIdx.x;
  if (e < Ee) unsafeAtomicAdd(&deg[dst[e]], ew[e]);
}

__global__ void k_dinv(float* __restrict__ deg) {
  int i = blockIdx.x * blockDim.x + threadIdx.x;
  if (i < Nn) {
    float d = deg[i];
    deg[i] = (d > 0.0f) ? rsqrtf(d) : 0.0f;  // becomes dinv in-place
  }
}

__global__ void k_norm(float* __restrict__ norm, const float* __restrict__ dinv,
                       const int* __restrict__ src, const int* __restrict__ dst,
                       const float* __restrict__ ew) {
  int e = blockIdx.x * blockDim.x + threadIdx.x;
  if (e < Ee) norm[e] = dinv[src[e]] * ew[e] * dinv[dst[e]];
}

// ---------------- fp32 -> bf16 transposed weight conversion ----------------
// In:  W  [K=128][M] row-major f32.  Out: WT [M][128] row-major bf16.
template <int M>
__global__ void k_f2bf_t(const float* __restrict__ W, __bf16* __restrict__ WT) {
  int i = blockIdx.x * blockDim.x + threadIdx.x;  // i indexes WT flat: m*128 + k
  if (i < M * 128) {
    int m = i >> 7, k = i & 127;
    WT[i] = f2bf(W[k * M + m]);
  }
}

// ---------------- WMMA GEMM: H[N x M] = X[N x 128] * W[128 x M] ----------------
// 256 threads = 8 waves; each wave computes a 16-row strip across all M cols.
// A fragments loaded straight from global f32 (2x 32B runs per fragment) and
// converted to bf16 in registers (packed v_cvt_pk_bf16_f32 when available);
// B (weights) staged transposed in LDS so each fragment is one contiguous 32B
// v16bf load (-> ds_load_b128 pairs).
template <int M>
__global__ __launch_bounds__(256)
void k_gemm(const float* __restrict__ X, const __bf16* __restrict__ WT,
            float* __restrict__ H, int nrows) {
  __shared__ __bf16 wldsT[M * 128];  // [col][k] layout

  const int tid  = threadIdx.x;
  const int wave = tid >> 5;
  const int lane = tid & 31;

  // vectorized copy of transposed bf16 weights into LDS (16B chunks)
  {
    const uint4* s4 = (const uint4*)WT;
    uint4* d4 = (uint4*)wldsT;
    const int n4 = (M * 128) / 8;
    for (int i = tid; i < n4; i += 256) d4[i] = s4[i];
  }
  __syncthreads();

  const long r0 = ((long)blockIdx.x * 8 + wave) * 16;
  if (r0 >= (long)nrows) return;  // nrows % 16 == 0 here

  const int g  = lane >> 4;   // half-wave group
  const int rn = lane & 15;   // A row / B,C column within tile

  const float* xrow = X + (r0 + rn) * 128;

  v8f acc[M / 16];
  v8f zero = {};
#pragma unroll
  for (int c = 0; c < M / 16; ++c) acc[c] = zero;

#pragma unroll
  for (int kk = 0; kk < 4; ++kk) {  // K = 128 in steps of 32
    const int kbase = kk * 32;
    // A fragment (contiguous k order, matching B): lane holds row rn,
    // elements 0..7  -> K = kbase + 16g + i       (32B contiguous)
    // elements 8..15 -> K = kbase + 16g + 8 + i-8 (32B contiguous)
    v8f lo = *(const v8f*)(xrow + kbase + 16 * g);
    v8f hi = *(const v8f*)(xrow + kbase + 16 * g + 8);
    v16bf a;
    cvt8_bf16(lo, a, 0);
    cvt8_bf16(hi, a, 8);
#pragma unroll
    for (int c = 0; c < M / 16; ++c) {
      const int col = c * 16 + rn;
      // B fragment: column col, K = kbase + 16g .. +15 -> contiguous 32B
      v16bf b = *(const v16bf*)&wldsT[col * 128 + kbase + 16 * g];
      acc[c] = __builtin_amdgcn_wmma_f32_16x16x32_bf16(
          false, a, false, b, (short)0, acc[c], false, false);
    }
  }

  // C/D layout: lanes 0-15 -> M=j, lanes 16-31 -> M=8+j; N = lane&15
#pragma unroll
  for (int c = 0; c < M / 16; ++c) {
    const int col = c * 16 + rn;
#pragma unroll
    for (int j = 0; j < 8; ++j) {
      long row = r0 + j + 8 * g;
      H[row * M + col] = acc[c][j];
    }
  }
}

// ---------------- aggregation: out = b + selfnorm*h  then += norm[e]*h[src] ----------------
template <int M>
__global__ void k_agg_init(float* __restrict__ out, const float* __restrict__ H,
                           const float* __restrict__ dinv, const float* __restrict__ bias) {
  long i = (long)blockIdx.x * blockDim.x + threadIdx.x;
  if (i >= (long)Nn * M) return;
  int m = (int)(i % M);
  long n = i / M;
  float di = dinv[n];
  out[i] = bias[m] + di * di * H[i];
}

template <int M>
__global__ __launch_bounds__(256)
void k_agg_edges(float* __restrict__ out, const float* __restrict__ H,
                 const float* __restrict__ norm, const int* __restrict__ src,
                 const int* __restrict__ dst) {
  int wid  = (blockIdx.x * blockDim.x + threadIdx.x) >> 5;  // one wave per edge
  int lane = threadIdx.x & 31;
  if (wid >= Ee) return;
  int s = src[wid], d = dst[wid];
  float w = norm[wid];
  const float* hs = H + (long)s * M;
  float* od = out + (long)d * M;
  if (lane == 0 && wid + 1 < Ee) {
    __builtin_prefetch(H + (long)src[wid + 1] * M, 0, 1);  // global_prefetch
  }
#pragma unroll
  for (int j = 0; j < M / 32; ++j) {
    unsafeAtomicAdd(od + lane + 32 * j, w * hs[lane + 32 * j]);
  }
}

// ---------------- batch norm + relu ----------------
__global__ void k_zero(float* __restrict__ p, int n) {
  int i = blockIdx.x * blockDim.x + threadIdx.x;
  if (i < n) p[i] = 0.0f;
}

__global__ __launch_bounds__(256)
void k_bn_stats(const float* __restrict__ H, float* __restrict__ stats) {
  int col  = threadIdx.x & 127;
  int rsub = threadIdx.x >> 7;  // 2 rows per block step
  float s = 0.0f, s2 = 0.0f;
  for (long r = (long)blockIdx.x * 2 + rsub; r < Nn; r += (long)gridDim.x * 2) {
    float v = H[r * 128 + col];
    s += v; s2 += v * v;
  }
  unsafeAtomicAdd(&stats[col], s);
  unsafeAtomicAdd(&stats[128 + col], s2);
}

__global__ void k_bn_apply(float* __restrict__ H, const float* __restrict__ stats,
                           const float* __restrict__ gamma, const float* __restrict__ beta) {
  long i = (long)blockIdx.x * blockDim.x + threadIdx.x;
  if (i >= (long)Nn * 128) return;
  int c = (int)(i & 127);
  const float invN = 1.0f / (float)Nn;
  float mean = stats[c] * invN;
  float var  = stats[128 + c] * invN - mean * mean;
  float v = (H[i] - mean) * rsqrtf(var + EPSS) * gamma[c] + beta[c];
  H[i] = v > 0.0f ? v : 0.0f;
}

// ---------------- host driver ----------------
extern "C" void kernel_launch(void* const* d_in, const int* in_sizes, int n_in,
                              void* d_out, int out_size, void* d_ws, size_t ws_size,
                              hipStream_t stream) {
  const float* x   = (const float*)d_in[0];
  const int*   ei  = (const int*)d_in[1];
  const int*   src = ei;
  const int*   dst = ei + Ee;
  const float* ew  = (const float*)d_in[2];
  const float* W1  = (const float*)d_in[3];
  const float* b1  = (const float*)d_in[4];
  const float* W2  = (const float*)d_in[5];
  const float* b2  = (const float*)d_in[6];
  const float* W3  = (const float*)d_in[7];
  const float* b3  = (const float*)d_in[8];
  const float* g1  = (const float*)d_in[9];
  const float* be1 = (const float*)d_in[10];
  const float* g2  = (const float*)d_in[11];
  const float* be2 = (const float*)d_in[12];
  float* out = (float*)d_out;

  // workspace carve-up (~55 MB)
  char* p = (char*)d_ws;
  float*  dinv  = (float*)p;  p += (size_t)65536 * 4;
  float*  norm  = (float*)p;  p += (size_t)Ee * 4;
  float*  bufA  = (float*)p;  p += (size_t)Nn * 128 * 4;
  float*  bufB  = (float*)p;  p += (size_t)Nn * 128 * 4;
  float*  stats = (float*)p;  p += 1024;
  __bf16* wt1   = (__bf16*)p; p += (size_t)128 * 128 * 2;
  __bf16* wt2   = (__bf16*)p; p += (size_t)128 * 128 * 2;
  __bf16* wt3   = (__bf16*)p; p += (size_t)64 * 128 * 2;

  const int TB = 256;
  const int gN  = (Nn + TB - 1) / TB;
  const int gE  = (Ee + TB - 1) / TB;
  const int gNM = ((long)Nn * 128 + TB - 1) / TB;
  const int gNO = ((long)Nn * 64 + TB - 1) / TB;
  const int gEdgeWaves = (Ee * 32 + TB - 1) / TB;  // one wave per edge
  const int gGemm = (Nn + 127) / 128;              // 8 row-tiles per block

  // normalization precompute
  k_deg_init<<<gN, TB, 0, stream>>>(dinv);
  k_deg_acc <<<gE, TB, 0, stream>>>(dinv, dst, ew);
  k_dinv    <<<gN, TB, 0, stream>>>(dinv);
  k_norm    <<<gE, TB, 0, stream>>>(norm, dinv, src, dst, ew);

  // weights -> transposed bf16
  k_f2bf_t<128><<<(128 * 128 + TB - 1) / TB, TB, 0, stream>>>(W1, wt1);
  k_f2bf_t<128><<<(128 * 128 + TB - 1) / TB, TB, 0, stream>>>(W2, wt2);
  k_f2bf_t<64> <<<(64 * 128 + TB - 1) / TB, TB, 0, stream>>>(W3, wt3);

  // ---- layer 1 ----
  k_gemm<128><<<gGemm, TB, 0, stream>>>(x, wt1, bufA, Nn);
  k_agg_init<128><<<gNM, TB, 0, stream>>>(bufB, bufA, dinv, b1);
  k_agg_edges<128><<<gEdgeWaves, TB, 0, stream>>>(bufB, bufA, norm, src, dst);
  k_zero<<<1, TB, 0, stream>>>(stats, 256);
  k_bn_stats<<<512, TB, 0, stream>>>(bufB, stats);
  k_bn_apply<<<gNM, TB, 0, stream>>>(bufB, stats, g1, be1);

  // ---- layer 2 ----
  k_gemm<128><<<gGemm, TB, 0, stream>>>(bufB, wt2, bufA, Nn);
  k_agg_init<128><<<gNM, TB, 0, stream>>>(bufB, bufA, dinv, b2);
  k_agg_edges<128><<<gEdgeWaves, TB, 0, stream>>>(bufB, bufA, norm, src, dst);
  k_zero<<<1, TB, 0, stream>>>(stats, 256);
  k_bn_stats<<<512, TB, 0, stream>>>(bufB, stats);
  k_bn_apply<<<gNM, TB, 0, stream>>>(bufB, stats, g2, be2);

  // ---- layer 3 (output, M=64) ----
  k_gemm<64><<<gGemm, TB, 0, stream>>>(bufB, wt3, bufA, Nn);
  k_agg_init<64><<<gNO, TB, 0, stream>>>(out, bufA, dinv, b3);
  k_agg_edges<64><<<gEdgeWaves, TB, 0, stream>>>(out, bufA, norm, src, dst);
}
